// MDTA_12051678232841
// MI455X (gfx1250) — compile-verified, matrix operand order checked by reading, not run on hardware
//
#include <hip/hip_runtime.h>
#include <hip/hip_bf16.h>
#include <math.h>

// ---------------------------------------------------------------------------
// MDTA with deformable-conv KV path, CDNA5 (gfx1250) wave32 WMMA pipeline.
// All matrix math runs through v_wmma_f32_16x16x32_bf16 with f32 accumulate.
// bf16 conversion uses hardware packed converts (v_cvt_pk_bf16_f32) via
// native casts / __builtin_convertvector instead of software rounding.
// ---------------------------------------------------------------------------

typedef __attribute__((ext_vector_type(16))) __bf16 v16bf;
typedef __attribute__((ext_vector_type(2)))  __bf16 v2bf;
typedef __attribute__((ext_vector_type(8)))  float  v8f;
typedef __attribute__((ext_vector_type(2)))  float  v2f;

#define HW     128
#define NPIX   16384   // 128*128
#define C1     192
#define C2     384
#define NBATCH 2
#define NHEAD  4
#define DHEAD  48

// ---- helpers --------------------------------------------------------------

__device__ __forceinline__ __bf16 f2bf(float x) {
  return (__bf16)x;                       // hardware f32->bf16 convert
}

__device__ __forceinline__ v2bf f2bf2(float a, float b) {
  v2f t; t[0] = a; t[1] = b;
  return __builtin_convertvector(t, v2bf);  // v_cvt_pk_bf16_f32
}

__device__ __forceinline__ int iclamp(int v, int lo, int hi) {
  return v < lo ? lo : (v > hi ? hi : v);
}

// K index held by VGPR slot v (0..7) for a 16-bit A/B operand, lane-half kh.
// Per ISA 7.12.2: V0..V3 hold K pairs {0,1..6,7} (lanes 0-15) / {8,9..14,15}
// (lanes 16-31), V4..V7 the same pattern +16.
__device__ __forceinline__ int kmap(int v, int khalf) {
  int base = (v < 4) ? (2 * v) : (16 + 2 * (v - 4));
  return base + 8 * khalf;
}

// A fragment (16x32), element (row, k) at src[row*row_stride + k*k_stride]
__device__ __forceinline__ v16bf load_a_f32(const float* src, int row_stride, int k_stride) {
  int lane = threadIdx.x & 31;
  int row = lane & 15, kh = lane >> 4;
  v16bf a;
#pragma unroll
  for (int v = 0; v < 8; ++v) {
    int k = kmap(v, kh);
    v2bf p = f2bf2(src[(size_t)row * row_stride + (size_t)k * k_stride],
                   src[(size_t)row * row_stride + (size_t)(k + 1) * k_stride]);
    a[2 * v]     = p[0];
    a[2 * v + 1] = p[1];
  }
  return a;
}

__device__ __forceinline__ v16bf load_a_bf(const __bf16* src, int row_stride, int k_stride) {
  int lane = threadIdx.x & 31;
  int row = lane & 15, kh = lane >> 4;
  v16bf a;
#pragma unroll
  for (int v = 0; v < 8; ++v) {
    int k = kmap(v, kh);
    a[2 * v]     = src[(size_t)row * row_stride + (size_t)k * k_stride];
    a[2 * v + 1] = src[(size_t)row * row_stride + (size_t)(k + 1) * k_stride];
  }
  return a;
}

// B fragment (32x16), element (k, n) at src[k*k_stride + n*n_stride]
__device__ __forceinline__ v16bf load_b_f32(const float* src, int k_stride, int n_stride) {
  int lane = threadIdx.x & 31;
  int col = lane & 15, kh = lane >> 4;
  v16bf b;
#pragma unroll
  for (int v = 0; v < 8; ++v) {
    int k = kmap(v, kh);
    v2bf p = f2bf2(src[(size_t)k * k_stride + (size_t)col * n_stride],
                   src[(size_t)(k + 1) * k_stride + (size_t)col * n_stride]);
    b[2 * v]     = p[0];
    b[2 * v + 1] = p[1];
  }
  return b;
}

__device__ __forceinline__ v16bf load_b_bf(const __bf16* src, int k_stride, int n_stride) {
  int lane = threadIdx.x & 31;
  int col = lane & 15, kh = lane >> 4;
  v16bf b;
#pragma unroll
  for (int v = 0; v < 8; ++v) {
    int k = kmap(v, kh);
    b[2 * v]     = src[(size_t)k * k_stride + (size_t)col * n_stride];
    b[2 * v + 1] = src[(size_t)(k + 1) * k_stride + (size_t)col * n_stride];
  }
  return b;
}

__device__ __forceinline__ v8f wmma_bf(v16bf a, v16bf b, v8f c) {
  return __builtin_amdgcn_wmma_f32_16x16x32_bf16(false, a, false, b, (short)0, c,
                                                 false, false);
}

// C/D 16x16 f32: VGPR r -> M = r + 8*(lane>=16), N = lane&15
__device__ __forceinline__ void store_c(float* dst, int ldc, v8f c) {
  int lane = threadIdx.x & 31;
  int n = lane & 15, mh = lane >> 4;
#pragma unroll
  for (int r = 0; r < 8; ++r)
    dst[(size_t)(r + 8 * mh) * ldc + n] = c[r];
}

// ---- 1. & 7.  Y[b] = W[MxK] @ X[b][KxN]  (1x1 convs) ----------------------
// grid: (N/16, M/64, B), block 128 (4 waves, one 16x16 M-tile each)
__global__ void gemm_wn_kernel(const float* __restrict__ W, const float* __restrict__ X,
                               float* __restrict__ Y, int M, int K) {
  int b  = blockIdx.z;
  int n0 = blockIdx.x * 16;
  int wv = threadIdx.x >> 5;
  int m0 = blockIdx.y * 64 + wv * 16;
  const float* Xb = X + (size_t)b * K * NPIX;
  v8f acc = {};
  for (int kc = 0; kc < K; kc += 32) {
    v16bf a  = load_a_f32(W + (size_t)m0 * K + kc, K, 1);
    v16bf bf = load_b_f32(Xb + (size_t)kc * NPIX + n0, NPIX, 1);
    acc = wmma_bf(a, bf, acc);
  }
  store_c(Y + ((size_t)b * M + m0) * NPIX + n0, NPIX, acc);
}

// ---- 2. 3x3 conv producing 18 offset channels (VALU; too skinny for WMMA) -
__global__ void offset_conv_kernel(const float* __restrict__ kv,
                                   const float* __restrict__ w,
                                   const float* __restrict__ bias,
                                   float* __restrict__ out) {
  int idx = blockIdx.x * blockDim.x + threadIdx.x;
  int total = NBATCH * 18 * NPIX;
  if (idx >= total) return;
  int n  = idx % NPIX;
  int oc = (idx / NPIX) % 18;
  int b  = idx / (18 * NPIX);
  int y = n >> 7, x = n & 127;
  float acc = bias[oc];
  const float* kvb = kv + (size_t)b * C2 * NPIX;
  const float* wo  = w + (size_t)oc * C2 * 9;
  for (int c = 0; c < C2; ++c) {
    const float* img = kvb + (size_t)c * NPIX;
    const float* wc  = wo + c * 9;
#pragma unroll
    for (int kh = 0; kh < 3; ++kh) {
      int yy = y - 1 + kh;
      if (yy < 0 || yy >= HW) continue;
#pragma unroll
      for (int kw = 0; kw < 3; ++kw) {
        int xx = x - 1 + kw;
        if (xx < 0 || xx >= HW) continue;
        acc += img[yy * HW + xx] * wc[kh * 3 + kw];
      }
    }
  }
  out[idx] = acc;
}

// ---- 3. deformable 3x3 conv as implicit GEMM ------------------------------
// Block = (batch, 32-pixel tile); 8 waves. Per tap: bilinear-sample a
// 384(cin) x 32(px) bf16 slab into LDS once, then every wave runs WMMAs for
// its 3 output-channel tiles (24 tiles total = 384 out channels).
__global__ void deform_conv_kernel(const float* __restrict__ kv,
                                   const float* __restrict__ offset,
                                   const float* __restrict__ reg_w,
                                   float* __restrict__ out) {
  __shared__ __bf16 S[C2 * 32];                     // 24 KB sampled slab
  __shared__ float sw0[32], sw1[32], sw2[32], sw3[32];
  __shared__ int   sy0[32], sy1[32], sx0[32], sx1[32];

  int b  = blockIdx.y;
  int p0 = blockIdx.x * 32;
  int tid = threadIdx.x;
  int wv  = tid >> 5;

  v8f acc[3][2];
#pragma unroll
  for (int i = 0; i < 3; ++i)
#pragma unroll
    for (int j = 0; j < 2; ++j) { v8f z = {}; acc[i][j] = z; }

  const float* kvb = kv + (size_t)b * C2 * NPIX;

  for (int t = 0; t < 9; ++t) {
    int kh = t / 3, kw = t % 3;
    if (tid < 32) {
      int p = p0 + tid;
      int y = p >> 7, x = p & 127;
      float dy = offset[((size_t)b * 18 + 2 * t) * NPIX + p];
      float dx = offset[((size_t)b * 18 + 2 * t + 1) * NPIX + p];
      float py = (float)(y - 1 + kh) + dy;
      float px = (float)(x - 1 + kw) + dx;
      float fy = floorf(py), fx = floorf(px);
      int y0 = (int)fy, x0 = (int)fx;
      float ay = py - fy, ax = px - fx;
      bool vy0 = (y0 >= 0) && (y0 < HW);
      bool vy1 = (y0 + 1 >= 0) && (y0 + 1 < HW);
      bool vx0 = (x0 >= 0) && (x0 < HW);
      bool vx1 = (x0 + 1 >= 0) && (x0 + 1 < HW);
      float wy0 = 1.f - ay, wy1 = ay, wx0 = 1.f - ax, wx1 = ax;
      sw0[tid] = (vy0 && vx0) ? wy0 * wx0 : 0.f;
      sw1[tid] = (vy0 && vx1) ? wy0 * wx1 : 0.f;
      sw2[tid] = (vy1 && vx0) ? wy1 * wx0 : 0.f;
      sw3[tid] = (vy1 && vx1) ? wy1 * wx1 : 0.f;
      sy0[tid] = iclamp(y0, 0, HW - 1);
      sy1[tid] = iclamp(y0 + 1, 0, HW - 1);
      sx0[tid] = iclamp(x0, 0, HW - 1);
      sx1[tid] = iclamp(x0 + 1, 0, HW - 1);
    }
    __syncthreads();

    // cooperative bilinear gather: 384*32 samples / 256 threads
    for (int i = tid; i < C2 * 32; i += 256) {
      int c = i >> 5, p = i & 31;
      const float* img = kvb + (size_t)c * NPIX;
      float val = sw0[p] * img[sy0[p] * HW + sx0[p]]
                + sw1[p] * img[sy0[p] * HW + sx1[p]]
                + sw2[p] * img[sy1[p] * HW + sx0[p]]
                + sw3[p] * img[sy1[p] * HW + sx1[p]];
      S[c * 32 + p] = f2bf(val);
    }
    __syncthreads();

    // implicit GEMM over cin for this tap; A = reg_w[o, cin, kh, kw]
    const float* wbase = reg_w + kh * 3 + kw;
    for (int kc = 0; kc < C2; kc += 32) {
      v16bf b0 = load_b_bf(S + kc * 32, 32, 1);
      v16bf b1 = load_b_bf(S + kc * 32 + 16, 32, 1);
#pragma unroll
      for (int i = 0; i < 3; ++i) {
        int m0 = (wv * 3 + i) * 16;
        v16bf af = load_a_f32(wbase + ((size_t)m0 * C2 + kc) * 9, C2 * 9, 9);
        acc[i][0] = wmma_bf(af, b0, acc[i][0]);
        acc[i][1] = wmma_bf(af, b1, acc[i][1]);
      }
    }
    __syncthreads();  // protect LDS before next tap's sampling
  }

#pragma unroll
  for (int i = 0; i < 3; ++i) {
    int m0 = (wv * 3 + i) * 16;
    store_c(out + ((size_t)b * C2 + m0) * NPIX + p0,      NPIX, acc[i][0]);
    store_c(out + ((size_t)b * C2 + m0) * NPIX + p0 + 16, NPIX, acc[i][1]);
  }
}

// ---- 4. per-row L2 normalize (f32 reduce) + bf16 cast ---------------------
// one 256-thread block per row of length NPIX
__global__ void rownorm_kernel(const float* __restrict__ src, int src_rows_per_b,
                               int c_off, float* __restrict__ dst_f32,
                               __bf16* __restrict__ dst_bf, int do_norm) {
  int r = blockIdx.x;            // 0 .. B*192-1
  int b = r / C1, c = r % C1;
  const float* row = src + ((size_t)b * src_rows_per_b + c_off + c) * NPIX;
  __shared__ float red[256];
  float s = 0.f;
  for (int i = threadIdx.x; i < NPIX; i += 256) { float v = row[i]; s += v * v; }
  red[threadIdx.x] = s;
  __syncthreads();
  for (int off = 128; off > 0; off >>= 1) {
    if (threadIdx.x < off) red[threadIdx.x] += red[threadIdx.x + off];
    __syncthreads();
  }
  float scale = 1.f;
  if (do_norm) scale = 1.f / fmaxf(sqrtf(red[0]), 1e-12f);
  for (int i = threadIdx.x; i < NPIX; i += 256) {
    float v = row[i] * scale;
    if (dst_f32) dst_f32[(size_t)r * NPIX + i] = v;
    dst_bf[(size_t)r * NPIX + i] = f2bf(v);
  }
}

// ---- 5. attention scores (48x48, K=16384) + softmax, one wave per (b,h) ---
__global__ void attn_softmax_kernel(const __bf16* __restrict__ qbf,
                                    const __bf16* __restrict__ kbf,
                                    const float* __restrict__ temperature,
                                    __bf16* __restrict__ attn_bf) {
  int bh = blockIdx.x;
  int h  = bh & (NHEAD - 1);
  const __bf16* Q  = qbf + (size_t)bh * DHEAD * NPIX;
  const __bf16* Km = kbf + (size_t)bh * DHEAD * NPIX;

  v8f acc[3][3];
#pragma unroll
  for (int i = 0; i < 3; ++i)
#pragma unroll
    for (int j = 0; j < 3; ++j) { v8f z = {}; acc[i][j] = z; }

  for (int kc = 0; kc < NPIX; kc += 32) {
    v16bf a[3], bt[3];
#pragma unroll
    for (int i = 0; i < 3; ++i)
      a[i] = load_a_bf(Q + (size_t)(i * 16) * NPIX + kc, NPIX, 1);
#pragma unroll
    for (int j = 0; j < 3; ++j)                      // B[k][e] = K[e][k]
      bt[j] = load_b_bf(Km + (size_t)(j * 16) * NPIX + kc, 1, NPIX);
#pragma unroll
    for (int i = 0; i < 3; ++i)
#pragma unroll
      for (int j = 0; j < 3; ++j)
        acc[i][j] = wmma_bf(a[i], bt[j], acc[i][j]);
  }

  __shared__ float sc[DHEAD][DHEAD];
  float temp = temperature[h];
  int lane = threadIdx.x & 31;
  int n = lane & 15, mh = lane >> 4;
#pragma unroll
  for (int i = 0; i < 3; ++i)
#pragma unroll
    for (int j = 0; j < 3; ++j)
#pragma unroll
      for (int r = 0; r < 8; ++r)
        sc[i * 16 + r + 8 * mh][j * 16 + n] = acc[i][j][r] * temp;
  __syncthreads();

  for (int rr = threadIdx.x; rr < DHEAD; rr += 32) {
    float mx = -1e30f;
    for (int e = 0; e < DHEAD; ++e) mx = fmaxf(mx, sc[rr][e]);
    float ssum = 0.f;
    for (int e = 0; e < DHEAD; ++e) {
      float ex = __expf(sc[rr][e] - mx);
      sc[rr][e] = ex;
      ssum += ex;
    }
    float inv = 1.f / ssum;
    for (int e = 0; e < 64; ++e)    // pad K to 64 with zeros for next GEMM
      attn_bf[((size_t)bh * DHEAD + rr) * 64 + e] =
          f2bf(e < DHEAD ? sc[rr][e] * inv : 0.f);
  }
}

// ---- 6. out = attn(48x48, padded K=64) @ V(48 x NPIX) ---------------------
// grid (NPIX/16, B*heads), one wave per block
__global__ void attn_v_kernel(const __bf16* __restrict__ attn_bf,
                              const __bf16* __restrict__ vbf,
                              float* __restrict__ out) {
  int bh = blockIdx.y;
  int n0 = blockIdx.x * 16;
  int lane = threadIdx.x & 31;
  int col = lane & 15, khalf = lane >> 4;

  v8f acc[3];
#pragma unroll
  for (int i = 0; i < 3; ++i) { v8f z = {}; acc[i] = z; }

  for (int kc = 0; kc < 64; kc += 32) {
    v16bf bfrag;                              // B[k=e][n]; clamp e>=48 (A is 0 there)
#pragma unroll
    for (int v = 0; v < 8; ++v) {
      int k = kmap(v, khalf) + kc;
      int e0 = k < DHEAD ? k : DHEAD - 1;
      int e1 = (k + 1) < DHEAD ? (k + 1) : DHEAD - 1;
      bfrag[2 * v]     = vbf[((size_t)bh * DHEAD + e0) * NPIX + n0 + col];
      bfrag[2 * v + 1] = vbf[((size_t)bh * DHEAD + e1) * NPIX + n0 + col];
    }
#pragma unroll
    for (int i = 0; i < 3; ++i) {
      v16bf a = load_a_bf(attn_bf + ((size_t)bh * DHEAD + i * 16) * 64 + kc, 64, 1);
      acc[i] = wmma_bf(a, bfrag, acc[i]);
    }
  }
  int b = bh >> 2, h = bh & 3;
#pragma unroll
  for (int i = 0; i < 3; ++i)
    store_c(out + ((size_t)b * C1 + h * DHEAD + i * 16) * NPIX + n0, NPIX, acc[i]);
}

// ---------------------------------------------------------------------------
extern "C" void kernel_launch(void* const* d_in, const int* in_sizes, int n_in,
                              void* d_out, int out_size, void* d_ws, size_t ws_size,
                              hipStream_t stream) {
  (void)in_sizes; (void)n_in; (void)out_size; (void)ws_size;
  const float* x      = (const float*)d_in[0];
  const float* q      = (const float*)d_in[1];
  const float* temp   = (const float*)d_in[2];
  const float* kv_w   = (const float*)d_in[3];
  const float* off_w  = (const float*)d_in[4];
  const float* off_b  = (const float*)d_in[5];
  const float* reg_w  = (const float*)d_in[6];
  const float* proj_w = (const float*)d_in[7];

  float* out    = (float*)d_out;
  float* qh_out = out + (size_t)NBATCH * C1 * NPIX;   // second tuple output

  // workspace layout (~166 MB total)
  float* ws_kv  = (float*)d_ws;                              // [B,384,N] f32
  float* ws_off = ws_kv  + (size_t)NBATCH * C2 * NPIX;       // [B,18,N]  f32
  float* ws_kvd = ws_off + (size_t)NBATCH * 18 * NPIX;       // [B,384,N] f32
  float* ws_ao  = ws_kvd + (size_t)NBATCH * C2 * NPIX;       // [B,192,N] f32
  __bf16* ws_qbf  = (__bf16*)(ws_ao + (size_t)NBATCH * C1 * NPIX);
  __bf16* ws_kbf  = ws_qbf + (size_t)NBATCH * C1 * NPIX;
  __bf16* ws_vbf  = ws_kbf + (size_t)NBATCH * C1 * NPIX;
  __bf16* ws_attn = ws_vbf + (size_t)NBATCH * C1 * NPIX;     // [B*H,48,64] bf16

  // 1. kv = kv_w @ x
  gemm_wn_kernel<<<dim3(NPIX / 16, C2 / 64, NBATCH), 128, 0, stream>>>(
      kv_w, x, ws_kv, C2, C1);

  // 2. offsets = conv3x3(kv)
  int tot = NBATCH * 18 * NPIX;
  offset_conv_kernel<<<(tot + 255) / 256, 256, 0, stream>>>(ws_kv, off_w, off_b, ws_off);

  // 3. deformable conv (implicit GEMM via LDS-staged bilinear samples)
  deform_conv_kernel<<<dim3(NPIX / 32, NBATCH), 256, 0, stream>>>(
      ws_kv, ws_off, reg_w, ws_kvd);

  // 4. l2norm q (also the second output), l2norm k, cast v
  rownorm_kernel<<<NBATCH * C1, 256, 0, stream>>>(q,      C1, 0,  qh_out,  ws_qbf, 1);
  rownorm_kernel<<<NBATCH * C1, 256, 0, stream>>>(ws_kvd, C2, 0,  nullptr, ws_kbf, 1);
  rownorm_kernel<<<NBATCH * C1, 256, 0, stream>>>(ws_kvd, C2, C1, nullptr, ws_vbf, 0);

  // 5. attn = softmax((q_n @ k_n^T) * temperature)
  attn_softmax_kernel<<<NBATCH * NHEAD, 32, 0, stream>>>(ws_qbf, ws_kbf, temp, ws_attn);

  // 6. attn @ v
  attn_v_kernel<<<dim3(NPIX / 16, NBATCH * NHEAD), 32, 0, stream>>>(ws_attn, ws_vbf, ws_ao);

  // 7. out = proj_w @ (attn @ v)
  gemm_wn_kernel<<<dim3(NPIX / 16, C1 / 64, NBATCH), 128, 0, stream>>>(
      proj_w, ws_ao, out, C1, C1);
}
